// PrototypeHead_19000935318141
// MI455X (gfx1250) — compile-verified, matrix operand order checked by reading, not run on hardware
//
#include <hip/hip_runtime.h>
#include <hip/hip_bf16.h>
#include <math.h>

// ---- problem constants (from reference) ----
#define BB 65536
#define DD 1024
#define CC 100
#define KK 5
#define KT 500          // C*K
#define KTPAD 512
#define NPAD 112        // 7 * 16 >= 100

typedef __bf16 bf16_t;
typedef __attribute__((ext_vector_type(16))) __bf16 v16bf;
typedef __attribute__((ext_vector_type(8)))  __bf16 v8bf;
typedef __attribute__((ext_vector_type(8)))  float  v8f;

// ---------------------------------------------------------------------------
// helpers
// ---------------------------------------------------------------------------
__device__ inline float blockReduceSum(float v, float* sh) {
    const int lane = threadIdx.x & 31;
    const int wid  = threadIdx.x >> 5;
#pragma unroll
    for (int o = 16; o; o >>= 1) v += __shfl_xor(v, o, 32);
    if (lane == 0) sh[wid] = v;
    __syncthreads();
    const int nw = (blockDim.x + 31) >> 5;
    float r = (threadIdx.x < nw) ? sh[threadIdx.x] : 0.0f;
    if (wid == 0) {
#pragma unroll
        for (int o = 16; o; o >>= 1) r += __shfl_xor(r, o, 32);
        if (lane == 0) sh[0] = r;
    }
    __syncthreads();
    float out = sh[0];
    __syncthreads();
    return out;
}

__device__ inline v16bf cvt16(const float4& f0, const float4& f1,
                              const float4& f2, const float4& f3) {
    v16bf a;
    a[0]  = (bf16_t)f0.x; a[1]  = (bf16_t)f0.y; a[2]  = (bf16_t)f0.z; a[3]  = (bf16_t)f0.w;
    a[4]  = (bf16_t)f1.x; a[5]  = (bf16_t)f1.y; a[6]  = (bf16_t)f1.z; a[7]  = (bf16_t)f1.w;
    a[8]  = (bf16_t)f2.x; a[9]  = (bf16_t)f2.y; a[10] = (bf16_t)f2.z; a[11] = (bf16_t)f2.w;
    a[12] = (bf16_t)f3.x; a[13] = (bf16_t)f3.y; a[14] = (bf16_t)f3.z; a[15] = (bf16_t)f3.w;
    return a;
}

__device__ inline float sq16(const float4& f0, const float4& f1,
                             const float4& f2, const float4& f3) {
    return f0.x*f0.x + f0.y*f0.y + f0.z*f0.z + f0.w*f0.w
         + f1.x*f1.x + f1.y*f1.y + f1.z*f1.z + f1.w*f1.w
         + f2.x*f2.x + f2.y*f2.y + f2.z*f2.z + f2.w*f2.w
         + f3.x*f3.x + f3.y*f3.y + f3.z*f3.z + f3.w*f3.w;
}

// ---------------------------------------------------------------------------
// 1) prot_n = l2norm(prototypes)    [KT, D] fp32 -> ws
// ---------------------------------------------------------------------------
__global__ void __launch_bounds__(256) k_norm_prot(const float* __restrict__ prot,
                                                   float* __restrict__ protn) {
    __shared__ float sh[32];
    const int r = blockIdx.x;
    const float* p = prot + (size_t)r * DD;
    float s = 0.0f;
    for (int d = threadIdx.x; d < DD; d += 256) { float x = p[d]; s += x * x; }
    const float tot = blockReduceSum(s, sh);
    const float inv = 1.0f / fmaxf(sqrtf(tot), 1e-12f);
    float* o = protn + (size_t)r * DD;
    for (int d = threadIdx.x; d < DD; d += 256) o[d] = p[d] * inv;
}

// ---------------------------------------------------------------------------
// 2) w = softmax(agg/T)*K ; wp[c] = sum_k w[c,k]*prot_n[c*K+k]  -> bf16 [NPAD, D]
// ---------------------------------------------------------------------------
__global__ void __launch_bounds__(256) k_wp(const float* __restrict__ agg,
                                            const float* __restrict__ protn,
                                            bf16_t* __restrict__ wp) {
    const int c = blockIdx.x;
    bf16_t* o = wp + (size_t)c * DD;
    if (c >= CC) {  // zero padding rows so GEMM needs no N guards on loads
        for (int d = threadIdx.x; d < DD; d += 256) o[d] = (bf16_t)0.0f;
        return;
    }
    float a[KK], m = -1e30f;
#pragma unroll
    for (int k = 0; k < KK; ++k) { a[k] = agg[c * KK + k] * 10.0f; m = fmaxf(m, a[k]); } // /0.1
    float sum = 0.0f;
#pragma unroll
    for (int k = 0; k < KK; ++k) { a[k] = expf(a[k] - m); sum += a[k]; }
#pragma unroll
    for (int k = 0; k < KK; ++k) a[k] = a[k] / sum * (float)KK;

    for (int d = threadIdx.x; d < DD; d += 256) {
        float acc = 0.0f;
#pragma unroll
        for (int k = 0; k < KK; ++k) acc += a[k] * protn[(size_t)(c * KK + k) * DD + d];
        o[d] = (bf16_t)acc;
    }
}

// ---------------------------------------------------------------------------
// 3) class_logits = (features @ wpT) / ||features||   via bf16 WMMA
//    block = 256 thr = 8 waves; wave owns 16 rows x 112 cols (7 accumulators)
// ---------------------------------------------------------------------------
__global__ void __launch_bounds__(256) k_logits(const float* __restrict__ feat,
                                                const bf16_t* __restrict__ wp,
                                                float* __restrict__ out) {
    __shared__ float invSh[8][16];
    const int lane = threadIdx.x & 31;
    const int wv   = threadIdx.x >> 5;
    const int r0   = lane & 15;     // A row / B col within tile
    const int half = lane >> 4;     // K-half selector (wave32 layout)
    const int rowBase = blockIdx.x * 128 + wv * 16;
    const float* aRow = feat + (size_t)(rowBase + r0) * DD;

    v8f acc[7];
#pragma unroll
    for (int t = 0; t < 7; ++t)
#pragma unroll
        for (int e = 0; e < 8; ++e) acc[t][e] = 0.0f;

    float s = 0.0f;
    for (int kb = 0; kb < DD; kb += 32) {
        const int kg0 = kb + half * 8;
        const int kg1 = kb + 16 + half * 8;
        __builtin_prefetch(aRow + kb + 64, 0, 0);   // global_prefetch_b8
        const float4 f0 = *(const float4*)(aRow + kg0);
        const float4 f1 = *(const float4*)(aRow + kg0 + 4);
        const float4 f2 = *(const float4*)(aRow + kg1);
        const float4 f3 = *(const float4*)(aRow + kg1 + 4);
        s += sq16(f0, f1, f2, f3);                  // each (row,k) counted exactly once
        const v16bf a = cvt16(f0, f1, f2, f3);
#pragma unroll
        for (int t = 0; t < 7; ++t) {
            const bf16_t* wRow = wp + (size_t)(t * 16 + r0) * DD;
            const v8bf b0 = *(const v8bf*)(wRow + kg0);
            const v8bf b1 = *(const v8bf*)(wRow + kg1);
            v16bf b;
#pragma unroll
            for (int e = 0; e < 8; ++e) { b[e] = b0[e]; b[8 + e] = b1[e]; }
            acc[t] = __builtin_amdgcn_wmma_f32_16x16x32_bf16(
                false, a, false, b, (short)0, acc[t], false, false);
        }
    }

    const float tot = s + __shfl_xor(s, 16, 32);    // combine the two K-halves (wave32)
    if (lane < 16) invSh[wv][lane] = 1.0f / fmaxf(sqrtf(tot), 1e-12f);
    __syncthreads();

#pragma unroll
    for (int t = 0; t < 7; ++t) {
        const int n = t * 16 + r0;
        if (n < CC) {
#pragma unroll
            for (int r = 0; r < 8; ++r) {
                const int row = rowBase + r + 8 * half;   // C/D layout: M = r + 8*(lane/16)
                out[(size_t)row * CC + n] = acc[t][r] * invSh[wv][r + 8 * half];
            }
        }
    }
}

// ---------------------------------------------------------------------------
// 4) sim = prot_n @ prot_nT  [KT,KT] via bf16 WMMA tiles (512-padded)
// ---------------------------------------------------------------------------
__global__ void __launch_bounds__(256) k_sim(const float* __restrict__ protn,
                                             float* __restrict__ sim) {
    const int lane = threadIdx.x & 31;
    const int wv   = threadIdx.x >> 5;
    const int r0   = lane & 15;
    const int half = lane >> 4;
    const int tile = blockIdx.x * 8 + wv;       // 1024 tiles = 32x32
    const int ti = tile >> 5, tj = tile & 31;
    const int ri = ti * 16 + r0;
    const int rj = tj * 16 + r0;
    const bool aOk = ri < KT, bOk = rj < KT;
    const float* aRow = protn + (size_t)ri * DD;
    const float* bRow = protn + (size_t)rj * DD;

    v8f acc;
#pragma unroll
    for (int e = 0; e < 8; ++e) acc[e] = 0.0f;

    const float4 z4 = make_float4(0.f, 0.f, 0.f, 0.f);
    for (int kb = 0; kb < DD; kb += 32) {
        const int kg0 = kb + half * 8;
        const int kg1 = kb + 16 + half * 8;
        float4 a0 = z4, a1 = z4, a2 = z4, a3 = z4;
        float4 b0 = z4, b1 = z4, b2 = z4, b3 = z4;
        if (aOk) {
            a0 = *(const float4*)(aRow + kg0); a1 = *(const float4*)(aRow + kg0 + 4);
            a2 = *(const float4*)(aRow + kg1); a3 = *(const float4*)(aRow + kg1 + 4);
        }
        if (bOk) {
            b0 = *(const float4*)(bRow + kg0); b1 = *(const float4*)(bRow + kg0 + 4);
            b2 = *(const float4*)(bRow + kg1); b3 = *(const float4*)(bRow + kg1 + 4);
        }
        const v16bf a = cvt16(a0, a1, a2, a3);
        const v16bf b = cvt16(b0, b1, b2, b3);
        acc = __builtin_amdgcn_wmma_f32_16x16x32_bf16(
            false, a, false, b, (short)0, acc, false, false);
    }

#pragma unroll
    for (int r = 0; r < 8; ++r) {
        const int i = ti * 16 + r + 8 * half;
        const int j = tj * 16 + r0;
        if (i < KT && j < KT) sim[(size_t)i * KT + j] = acc[r];
    }
}

// ---------------------------------------------------------------------------
// 5) Q: per-row grouped softmax (temp 0.05) + faithful 1+3 row renormalizations
// ---------------------------------------------------------------------------
__global__ void __launch_bounds__(128) k_q(const float* __restrict__ sim,
                                           float* __restrict__ Q) {
    __shared__ float sh[32];
    const int i = blockIdx.x;
    const float* srow = sim + (size_t)i * KT;
    float* qrow = Q + (size_t)i * KT;
    const int g = threadIdx.x;

    float e[KK];
    float gsum = 0.0f;
    if (g < CC) {
        float v[KK], m = -1e30f;
#pragma unroll
        for (int k = 0; k < KK; ++k) { v[k] = srow[g * KK + k] * 20.0f; m = fmaxf(m, v[k]); } // /0.05
        float ssum = 0.0f;
#pragma unroll
        for (int k = 0; k < KK; ++k) { e[k] = expf(v[k] - m); ssum += e[k]; }
        gsum = 0.0f;
#pragma unroll
        for (int k = 0; k < KK; ++k) { e[k] /= ssum; gsum += e[k]; }
    }
    float tot = blockReduceSum(gsum, sh);   // row sum over all 500 entries
    // Q /= (rowsum + eps), then 3 more iterations of the same (reference-faithful)
    float scale = 1.0f, r = tot;
#pragma unroll
    for (int it = 0; it < 4; ++it) { scale /= (r + 1e-6f); r = r / (r + 1e-6f); }
    if (g < CC) {
#pragma unroll
        for (int k = 0; k < KK; ++k) qrow[g * KK + k] = e[k] * scale;
    }
}

// ---------------------------------------------------------------------------
// 6) mass[j] = sum_i Q[i][j] + eps
// ---------------------------------------------------------------------------
__global__ void __launch_bounds__(256) k_mass(const float* __restrict__ Q,
                                              float* __restrict__ mass) {
    const int j = blockIdx.x * 256 + threadIdx.x;
    if (j >= KT) return;
    float s = 0.0f;
    for (int i = 0; i < KT; ++i) s += Q[(size_t)i * KT + j];
    mass[j] = s + 1e-6f;
}

// ---------------------------------------------------------------------------
// 7) U = EMA_M*prototypes + (1-EMA_M)*(QT @ prot_n)/mass
// ---------------------------------------------------------------------------
__global__ void __launch_bounds__(256) k_emaupd(const float* __restrict__ prot,
                                                const float* __restrict__ protn,
                                                const float* __restrict__ Q,
                                                const float* __restrict__ mass,
                                                float* __restrict__ U) {
    const int j = blockIdx.x;
    const float mj = mass[j];
    for (int d = threadIdx.x; d < DD; d += 256) {
        float s = 0.0f;
        for (int i = 0; i < KT; ++i) s += Q[(size_t)i * KT + j] * protn[(size_t)i * DD + d];
        U[(size_t)j * DD + d] = 0.99f * prot[(size_t)j * DD + d] + 0.01f * (s / mj);
    }
}

// ---------------------------------------------------------------------------
// 8) sequential modified Gram-Schmidt (reference-faithful), one resident block
// ---------------------------------------------------------------------------
__global__ void __launch_bounds__(1024) k_gs(float* __restrict__ U) {
    __shared__ float red[32];
    __shared__ float dotSh;
    const int tid = threadIdx.x, lane = tid & 31, wid = tid >> 5;
    for (int i = 1; i < KT; ++i) {
        float u = U[(size_t)i * DD + tid];
        for (int j = 0; j < i; ++j) {
            const float vj = U[(size_t)j * DD + tid];
            float p = u * vj;
#pragma unroll
            for (int o = 16; o; o >>= 1) p += __shfl_xor(p, o, 32);
            if (lane == 0) red[wid] = p;
            __syncthreads();
            if (wid == 0) {
                float q = red[lane];
#pragma unroll
                for (int o = 16; o; o >>= 1) q += __shfl_xor(q, o, 32);
                if (lane == 0) dotSh = q;
            }
            __syncthreads();
            u -= dotSh * vj;
        }
        U[(size_t)i * DD + tid] = u;
        __threadfence_block();
        __syncthreads();
    }
}

// ---------------------------------------------------------------------------
// 9) new_prototypes = l2norm(U) -> out
// ---------------------------------------------------------------------------
__global__ void __launch_bounds__(256) k_final(const float* __restrict__ U,
                                               float* __restrict__ out2) {
    __shared__ float sh[32];
    const int r = blockIdx.x;
    const float* p = U + (size_t)r * DD;
    float s = 0.0f;
    for (int d = threadIdx.x; d < DD; d += 256) { float x = p[d]; s += x * x; }
    const float tot = blockReduceSum(s, sh);
    const float inv = 1.0f / fmaxf(sqrtf(tot), 1e-12f);
    for (int d = threadIdx.x; d < DD; d += 256) out2[(size_t)r * DD + d] = p[d] * inv;
}

// ---------------------------------------------------------------------------
// launcher
// ---------------------------------------------------------------------------
extern "C" void kernel_launch(void* const* d_in, const int* in_sizes, int n_in,
                              void* d_out, int out_size, void* d_ws, size_t ws_size,
                              hipStream_t stream) {
    const float* features   = (const float*)d_in[0];   // [B, D]
    const float* prototypes = (const float*)d_in[1];   // [KT, D]
    const float* agg        = (const float*)d_in[2];   // [C, K]

    float* out       = (float*)d_out;
    float* logitsOut = out;                                  // [B, C]
    float* protoOut  = out + (size_t)BB * CC;                // [KT, D]

    // workspace layout (floats)
    float*  ws    = (float*)d_ws;
    float*  protn = ws;                                      // 512000
    bf16_t* wp    = (bf16_t*)(ws + (size_t)KT * DD);         // NPAD*DD bf16 = 57344 fl
    float*  sim   = ws + (size_t)KT * DD + (NPAD * DD) / 2;  // 250000
    float*  Q     = sim + (size_t)KT * KT;                   // 250000
    float*  mass  = Q + (size_t)KT * KT;                     // 512
    float*  U     = mass + 512;                              // 512000

    k_norm_prot<<<KT, 256, 0, stream>>>(prototypes, protn);
    k_wp<<<NPAD, 256, 0, stream>>>(agg, protn, wp);
    k_logits<<<BB / 128, 256, 0, stream>>>(features, wp, logitsOut);
    k_sim<<<(32 * 32) / 8, 256, 0, stream>>>(protn, sim);
    k_q<<<KT, 128, 0, stream>>>(sim, Q);
    k_mass<<<2, 256, 0, stream>>>(Q, mass);
    k_emaupd<<<KT, 256, 0, stream>>>(prototypes, protn, Q, mass, U);
    k_gs<<<1, 1024, 0, stream>>>(U);
    k_final<<<KT, 256, 0, stream>>>(U, protoOut);
}
